// GCN_37125697307411
// MI455X (gfx1250) — matrix operand order repaired and checked by reading, not compiled, not run on hardware
//
#include <hip/hip_runtime.h>
#include <stdint.h>

typedef __attribute__((ext_vector_type(16))) __bf16       bf16x16;
typedef __attribute__((ext_vector_type(8)))  float        v8f;
typedef __attribute__((ext_vector_type(8)))  unsigned int u32x8;

#define NFEAT 512
#define NHID  128
#define NCLS  40
#define AP1B  520   // bf16 LDS row stride for 512-wide panel (1040B % 256 = 16 -> spread banks)
#define AP2B  136   // bf16 LDS row stride for 128-wide panel (272B  % 256 = 16 -> spread banks)

// ---- Kernel 0: one-time weight prep: bf16, column-major (K contiguous) ----
__global__ __launch_bounds__(256) void prep_w_kernel(const float* __restrict__ W1,
    const float* __restrict__ W2, __bf16* __restrict__ w1t, __bf16* __restrict__ w2t)
{
  const int i = blockIdx.x * blockDim.x + threadIdx.x;
  if (i < NFEAT * NHID) {            // w1t[n*512 + k] = W1[k*128 + n]
    const int n = i >> 9, k = i & 511;
    w1t[i] = (__bf16)W1[(size_t)k * NHID + n];
  }
  if (i < NCLS * NHID) {             // w2t[n*128 + k] = W2[k*40 + n]
    const int n = i >> 7, k = i & 127;
    w2t[i] = (__bf16)W2[(size_t)k * NCLS + n];
  }
}

// ---------------- Kernel 1: XW1 = X @ W1  (bf16 WMMA, f32 accum) ----------------
// 256 thr = 8 waves; block covers 64 rows x 128 cols; each wave: 16-col stripe,
// 4 row-tiles reusing one B fragment per K-step (4 WMMA / step, 16 steps).
__global__ __launch_bounds__(256) void gemm1_kernel(const float* __restrict__ x,
    const __bf16* __restrict__ w1t, float* __restrict__ xw1, int nrows)
{
  __shared__ unsigned AsU[(64 * AP1B) / 2];
  const int row0 = blockIdx.x * 64;
  const int tid  = threadIdx.x;
  // fill 64x512 panel as packed bf16 pairs (f32 -> bf16 done once here)
  for (int p = tid; p < 64 * 256; p += 256) {
    const int r  = p >> 8;
    const int c2 = (p & 255) * 2;
    const int gr = row0 + r;
    float f0 = 0.f, f1 = 0.f;
    if (gr < nrows) {
      const float2 v = *reinterpret_cast<const float2*>(x + (size_t)gr * NFEAT + c2);
      f0 = v.x; f1 = v.y;
    }
    const unsigned short l0 = __builtin_bit_cast(unsigned short, (__bf16)f0);
    const unsigned short l1 = __builtin_bit_cast(unsigned short, (__bf16)f1);
    AsU[(r * AP1B + c2) >> 1] = (unsigned)l0 | ((unsigned)l1 << 16);
  }
  __syncthreads();
  const int wave = tid >> 5, lane = tid & 31;
  const int col0 = wave * 16;
  const int mn   = lane & 15, half = lane >> 4;
  const unsigned* w1u = reinterpret_cast<const unsigned*>(w1t);
  v8f acc[4] = {{}, {}, {}, {}};
  for (int kb = 0; kb < NFEAT; kb += 32) {
    // B 32x16: lane n holds K = half*16 + 2i {+0,+1}; K-major bf16 -> 8 consecutive u32
    u32x8 bw;
#pragma unroll
    for (int i = 0; i < 8; ++i)
      bw[i] = w1u[(size_t)(col0 + mn) * 256 + (kb >> 1) + half * 8 + i];
    const bf16x16 b = __builtin_bit_cast(bf16x16, bw);
#pragma unroll
    for (int t = 0; t < 4; ++t) {
      // A 16x32: lane m holds K = half*16 + 2i {+0,+1}
      u32x8 aw;
#pragma unroll
      for (int i = 0; i < 8; ++i) {
        const int ka = half * 16 + i * 2;
        aw[i] = AsU[((t * 16 + mn) * AP1B + kb + ka) >> 1];
      }
      const bf16x16 a = __builtin_bit_cast(bf16x16, aw);
      acc[t] = __builtin_amdgcn_wmma_f32_16x16x32_bf16(false, a, false, b,
                                                       (short)0, acc[t], false, false);
    }
  }
  // C/D: VGPR i -> row half*8+i, col lane&15
  if (row0 + 64 <= nrows) {          // uniform fast path, no per-lane guards
#pragma unroll
    for (int t = 0; t < 4; ++t)
#pragma unroll
      for (int i = 0; i < 8; ++i)
        xw1[(size_t)(row0 + t * 16 + half * 8 + i) * NHID + col0 + mn] = acc[t][i];
  } else {
#pragma unroll
    for (int t = 0; t < 4; ++t)
#pragma unroll
      for (int i = 0; i < 8; ++i) {
        const int gr = row0 + t * 16 + half * 8 + i;
        if (gr < nrows) xw1[(size_t)gr * NHID + col0 + mn] = acc[t][i];
      }
  }
}

// ---------------- Kernel 2: init H accumulator to 0, logits to b2 ----------------
__global__ __launch_bounds__(256) void init_kernel(float* __restrict__ h,
    float* __restrict__ logits, const float* __restrict__ b2, int nrows)
{
  const long long i = (long long)blockIdx.x * blockDim.x + threadIdx.x;
  if (i < (long long)nrows * NHID) h[i] = 0.f;
  if (i < (long long)nrows * NCLS) logits[i] = b2[(int)(i % NCLS)];
}

// ---------------- Kernel 3: H += scatter-add over edges (d=128) ----------------
__global__ __launch_bounds__(256) void spmm1_kernel(const long long* __restrict__ ei,
    const float* __restrict__ ew, const float* __restrict__ xw1,
    float* __restrict__ h, int nedges)
{
  const long long gid = (long long)blockIdx.x * blockDim.x + threadIdx.x;
  const int e = (int)(gid >> 5);
  if (e >= nedges) return;
  const int lane = (int)(gid & 31);
  const int src = (int)ei[e];
  const int dst = (int)ei[(size_t)nedges + e];
  const float w = ew[e];
  const float4 v = *reinterpret_cast<const float4*>(xw1 + (size_t)src * NHID + lane * 4);
  float* p = h + (size_t)dst * NHID + lane * 4;
  atomicAdd(p + 0, w * v.x);
  atomicAdd(p + 1, w * v.y);
  atomicAdd(p + 2, w * v.z);
  atomicAdd(p + 3, w * v.w);
}

// ---------------- Kernel 4: HW2 = relu(H + b1) @ W2  (bf16 WMMA) ----------------
// 96 thr = 3 waves (cols 0..47, 40 valid); 64 rows/block, 4 row-tiles per wave.
__global__ __launch_bounds__(96) void gemm2_kernel(const float* __restrict__ h,
    const float* __restrict__ b1, const __bf16* __restrict__ w2t,
    float* __restrict__ hw2, int nrows)
{
  __shared__ unsigned AsU[(64 * AP2B) / 2];
  const int row0 = blockIdx.x * 64;
  const int tid  = threadIdx.x;
  for (int p = tid; p < 64 * 64; p += 96) {   // 64 rows x 64 f32-pairs
    const int r  = p >> 6;
    const int c2 = (p & 63) * 2;
    const int gr = row0 + r;
    float f0 = 0.f, f1 = 0.f;
    if (gr < nrows) {
      const float2 v = *reinterpret_cast<const float2*>(h + (size_t)gr * NHID + c2);
      f0 = v.x + b1[c2]; f1 = v.y + b1[c2 + 1];
    }
    f0 = f0 > 0.f ? f0 : 0.f;                 // fused bias + ReLU
    f1 = f1 > 0.f ? f1 : 0.f;
    const unsigned short l0 = __builtin_bit_cast(unsigned short, (__bf16)f0);
    const unsigned short l1 = __builtin_bit_cast(unsigned short, (__bf16)f1);
    AsU[(r * AP2B + c2) >> 1] = (unsigned)l0 | ((unsigned)l1 << 16);
  }
  __syncthreads();
  const int wave = tid >> 5, lane = tid & 31;
  const int col0 = wave * 16;
  const int mn   = lane & 15, half = lane >> 4;
  const int col  = col0 + mn;
  const unsigned* w2u = reinterpret_cast<const unsigned*>(w2t);
  v8f acc[4] = {{}, {}, {}, {}};
  for (int kb = 0; kb < NHID; kb += 32) {
    u32x8 bw = {};
    if (col < NCLS) {
#pragma unroll
      for (int i = 0; i < 8; ++i)
        bw[i] = w2u[(size_t)col * 64 + (kb >> 1) + half * 8 + i];
    }
    const bf16x16 b = __builtin_bit_cast(bf16x16, bw);
#pragma unroll
    for (int t = 0; t < 4; ++t) {
      u32x8 aw;
#pragma unroll
      for (int i = 0; i < 8; ++i) {
        const int ka = half * 16 + i * 2;
        aw[i] = AsU[((t * 16 + mn) * AP2B + kb + ka) >> 1];
      }
      const bf16x16 a = __builtin_bit_cast(bf16x16, aw);
      acc[t] = __builtin_amdgcn_wmma_f32_16x16x32_bf16(false, a, false, b,
                                                       (short)0, acc[t], false, false);
    }
  }
  if (col < NCLS) {
    if (row0 + 64 <= nrows) {
#pragma unroll
      for (int t = 0; t < 4; ++t)
#pragma unroll
        for (int i = 0; i < 8; ++i)
          hw2[(size_t)(row0 + t * 16 + half * 8 + i) * NCLS + col] = acc[t][i];
    } else {
#pragma unroll
      for (int t = 0; t < 4; ++t)
#pragma unroll
        for (int i = 0; i < 8; ++i) {
          const int gr = row0 + t * 16 + half * 8 + i;
          if (gr < nrows) hw2[(size_t)gr * NCLS + col] = acc[t][i];
        }
    }
  }
}

// ---------------- Kernel 5: logits += scatter-add over edges (d=40) ----------------
__global__ __launch_bounds__(256) void spmm2_kernel(const long long* __restrict__ ei,
    const float* __restrict__ ew, const float* __restrict__ hw2,
    float* __restrict__ logits, int nedges)
{
  const long long gid = (long long)blockIdx.x * blockDim.x + threadIdx.x;
  const int e = (int)(gid >> 5);
  if (e >= nedges) return;
  const int lane = (int)(gid & 31);
  const int src = (int)ei[e];
  const int dst = (int)ei[(size_t)nedges + e];
  const float w = ew[e];
  for (int c = lane; c < NCLS; c += 32)
    atomicAdd(logits + (size_t)dst * NCLS + c, w * hw2[(size_t)src * NCLS + c]);
}

// ---------------- Kernel 6: row-wise log_softmax (one wave32 per row) ----------------
__global__ __launch_bounds__(256) void logsoftmax_kernel(const float* __restrict__ logits,
    float* __restrict__ out, int nrows)
{
  const long long gid = (long long)blockIdx.x * blockDim.x + threadIdx.x;
  const int row = (int)(gid >> 5);
  if (row >= nrows) return;
  const int lane = (int)(gid & 31);
  const float v0 = logits[(size_t)row * NCLS + lane];
  const float v1 = (lane < 8) ? logits[(size_t)row * NCLS + 32 + lane] : -3.4e38f;
  float m = fmaxf(v0, v1);
#pragma unroll
  for (int off = 16; off > 0; off >>= 1) m = fmaxf(m, __shfl_xor(m, off, 32));
  float s = expf(v0 - m) + ((lane < 8) ? expf(v1 - m) : 0.f);
#pragma unroll
  for (int off = 16; off > 0; off >>= 1) s += __shfl_xor(s, off, 32);
  const float l = m + logf(s);
  out[(size_t)row * NCLS + lane] = v0 - l;
  if (lane < 8) out[(size_t)row * NCLS + 32 + lane] = v1 - l;
}

extern "C" void kernel_launch(void* const* d_in, const int* in_sizes, int n_in,
                              void* d_out, int out_size, void* d_ws, size_t ws_size,
                              hipStream_t stream) {
  const float*     x  = (const float*)d_in[0];
  const long long* ei = (const long long*)d_in[1];
  const float*     ew = (const float*)d_in[2];
  const float*     W1 = (const float*)d_in[3];
  const float*     b1 = (const float*)d_in[4];
  const float*     W2 = (const float*)d_in[5];
  const float*     b2 = (const float*)d_in[6];
  const int nrows  = in_sizes[0] / NFEAT;   // 50000
  const int nedges = in_sizes[2];           // 800000
  float* out = (float*)d_out;

  float*  xw1    = (float*)d_ws;                         // [nrows,128]
  float*  h      = xw1 + (size_t)nrows * NHID;           // [nrows,128]
  float*  hw2    = h   + (size_t)nrows * NHID;           // [nrows,40]
  float*  logits = hw2 + (size_t)nrows * NCLS;           // [nrows,40]
  __bf16* w1t    = (__bf16*)(logits + (size_t)nrows * NCLS);  // [128,512] bf16
  __bf16* w2t    = w1t + (size_t)NHID * NFEAT;                // [40,128] bf16

  prep_w_kernel<<<(NFEAT * NHID + 255) / 256, 256, 0, stream>>>(W1, W2, w1t, w2t);

  const int mblocks = (nrows + 63) / 64;
  gemm1_kernel<<<mblocks, 256, 0, stream>>>(x, w1t, xw1, nrows);

  const long long tot = (long long)nrows * NHID;
  init_kernel<<<(int)((tot + 255) / 256), 256, 0, stream>>>(h, logits, b2, nrows);

  const int eblocks = (int)(((long long)nedges * 32 + 255) / 256);
  spmm1_kernel<<<eblocks, 256, 0, stream>>>(ei, ew, xw1, h, nedges);

  gemm2_kernel<<<mblocks, 96, 0, stream>>>(h, b1, w2t, hw2, nrows);

  spmm2_kernel<<<eblocks, 256, 0, stream>>>(ei, ew, hw2, logits, nedges);

  const int sblocks = (int)(((long long)nrows * 32 + 255) / 256);
  logsoftmax_kernel<<<sblocks, 256, 0, stream>>>(logits, out, nrows);
}
